// NLBlock_50319836840645
// MI455X (gfx1250) — matrix operand order repaired and checked by reading, not compiled, not run on hardware
//
#include <hip/hip_runtime.h>
#include <hip/hip_bf16.h>

typedef __attribute__((ext_vector_type(16))) __bf16 v16bf;
typedef __attribute__((ext_vector_type(8)))  __bf16 v8bf;
typedef __attribute__((ext_vector_type(8)))  float  v8f;
typedef __attribute__((ext_vector_type(4)))  float  f4;

#define N_ROWS 8192
#define L_ROWS 8192
#define DIM    512
#define ATT_SCALE 0.04419417382415922f   // (1/512)^0.5
#define LN_EPS 1e-5f

// ---------------------------------------------------------------------------
// CDNA5 async global->LDS copy (GV mode): LDS[vdst + 0..15] = MEM[vaddr..+15].
// Tracked by ASYNCcnt; complete with s_wait_asynccnt before the barrier.
// ---------------------------------------------------------------------------
__device__ inline void async_ld16(unsigned int lds_byte_off, const void* gaddr) {
  asm volatile("global_load_async_to_lds_b128 %0, %1, off"
               :: "v"(lds_byte_off), "v"(gaddr) : "memory");
}

__device__ inline void wait_async0() {
#if __has_builtin(__builtin_amdgcn_s_wait_asynccnt)
  __builtin_amdgcn_s_wait_asynccnt(0);
#else
  asm volatile("s_wait_asynccnt 0" ::: "memory");
#endif
}

// ---------------------------------------------------------------------------
// WMMA fragment loaders from LDS (bf16, 16x16x32 shapes).
// A (16x32, MxK): lane l<16 -> row l, K 0..7 & 16..23 ; lane l>=16 -> row l-16,
// K 8..15 & 24..31.  (CDNA5 ISA "16-bit A-Matrix 16x32")
// B (32x16, KxN): lane n<16 -> column n, K 0..7 & 16..23 ; lane n+16 -> column n, K 8..15 & 24..31.
// Both become 16B ds_load_b128 pairs when `stride` (in bf16 elems) is a
// multiple of 8 and the base is 16B aligned.
// ---------------------------------------------------------------------------
__device__ inline v16bf load_a_frag(const __bf16* base, int stride, int lane) {
  const int r  = lane & 15;
  const int kh = (lane >> 4) * 8;
  const __bf16* p = base + r * stride + kh;
  v8bf lo = *(const v8bf*)(p);
  v8bf hi = *(const v8bf*)(p + 16);
  v16bf f;
#pragma unroll
  for (int i = 0; i < 8; ++i) { f[i] = lo[i]; f[i + 8] = hi[i]; }
  return f;
}

__device__ inline v16bf load_b_frag(const __bf16* base, int stride, int lane) {
  const int r  = lane & 15;          // output column within the 16-wide tile
  const int ko = (lane >> 4) * 8;    // K half
  const __bf16* p = base + r * stride + ko;
  v8bf lo = *(const v8bf*)(p);
  v8bf hi = *(const v8bf*)(p + 16);
  v16bf f;
#pragma unroll
  for (int i = 0; i < 8; ++i) { f[i] = lo[i]; f[i + 8] = hi[i]; }
  return f;
}

// ---------------------------------------------------------------------------
// C = A[M,512] @ W[512,512]^T + bias, optionally + residual.
// outb != nullptr : write bf16 (projection path)
// outb == nullptr : write fp32 = C + resid (final path)
// Block: 256 thr (8 waves), tile 128x128, k-step 32. Waves: 4 (m) x 2 (n),
// each wave computes a 32x64 patch = 2x4 WMMA tiles.
// ---------------------------------------------------------------------------
#define GLDK 40   // padded LDS k-stride (bf16 elems), 80B rows, 16B aligned

__global__ __launch_bounds__(256) void gemm512(
    const float* __restrict__ A, const float* __restrict__ W,
    const float* __restrict__ bias, const float* __restrict__ resid,
    __bf16* __restrict__ outb, float* __restrict__ outf)
{
  __shared__ __bf16 sA[128 * GLDK];
  __shared__ __bf16 sB[128 * GLDK];

  const int tid  = threadIdx.x;
  const int lane = tid & 31;
  const int wave = tid >> 5;
  const int wm = wave & 3;        // 0..3 -> 32-row group
  const int wn = wave >> 2;       // 0..1 -> 64-col group
  const int m0 = blockIdx.y * 128;
  const int n0 = blockIdx.x * 128;

  const v8f vzero = {0.f, 0.f, 0.f, 0.f, 0.f, 0.f, 0.f, 0.f};
  v8f acc[2][4];
#pragma unroll
  for (int i = 0; i < 2; ++i)
#pragma unroll
    for (int j = 0; j < 4; ++j) acc[i][j] = vzero;

  const int lrow = tid >> 3;        // 0..31
  const int lcol = (tid & 7) * 4;   // 0..28

  for (int kk = 0; kk < DIM; kk += 32) {
    __syncthreads();
#pragma unroll
    for (int r = lrow; r < 128; r += 32) {
      f4 va = *(const f4*)(A + (size_t)(m0 + r) * DIM + kk + lcol);
      f4 vb = *(const f4*)(W + (size_t)(n0 + r) * DIM + kk + lcol);
      __bf16* da = &sA[r * GLDK + lcol];
      __bf16* db = &sB[r * GLDK + lcol];
#pragma unroll
      for (int e = 0; e < 4; ++e) { da[e] = (__bf16)va[e]; db[e] = (__bf16)vb[e]; }
      if (kk + 32 < DIM) {   // prefetch next k-tile -> global_prefetch_b8
        __builtin_prefetch(A + (size_t)(m0 + r) * DIM + kk + 32 + lcol, 0, 3);
        __builtin_prefetch(W + (size_t)(n0 + r) * DIM + kk + 32 + lcol, 0, 3);
      }
    }
    __syncthreads();

    v16bf af[2], bfr[4];
#pragma unroll
    for (int mi = 0; mi < 2; ++mi)
      af[mi] = load_a_frag(&sA[(wm * 32 + mi * 16) * GLDK], GLDK, lane);
#pragma unroll
    for (int ni = 0; ni < 4; ++ni)
      bfr[ni] = load_b_frag(&sB[(wn * 64 + ni * 16) * GLDK], GLDK, lane);
#pragma unroll
    for (int mi = 0; mi < 2; ++mi)
#pragma unroll
      for (int ni = 0; ni < 4; ++ni)
        acc[mi][ni] = __builtin_amdgcn_wmma_f32_16x16x32_bf16(
            false, af[mi], false, bfr[ni], (short)0, acc[mi][ni], false, false);
  }

  // C/D layout: VGPR i -> M = i (lanes 0-15) or 8+i (lanes 16-31), N = lane&15
  const int rlo = (lane >> 4) * 8;
  const int cn  = lane & 15;
#pragma unroll
  for (int mi = 0; mi < 2; ++mi) {
#pragma unroll
    for (int ni = 0; ni < 4; ++ni) {
      const int col = n0 + wn * 64 + ni * 16 + cn;
      const float bv = bias[col];
#pragma unroll
      for (int i = 0; i < 8; ++i) {
        const int row = m0 + wm * 32 + mi * 16 + rlo + i;
        const float v = acc[mi][ni][i] + bv;
        if (outb) outb[(size_t)row * DIM + col] = (__bf16)v;
        else      outf[(size_t)row * DIM + col] = v + resid[(size_t)row * DIM + col];
      }
    }
  }
}

// ---------------------------------------------------------------------------
// Flash attention: O = softmax(Q K^T * scale) @ V, Q/K/V bf16, O fp32.
// Block = 256 thr (8 waves) handles 64 Q rows; loop over L in 64-row chunks.
// Wave w owns O columns [64w, 64w+64): 4x4 f32 accumulator tiles (128 VGPRs).
// Q/K are staged with CDNA5 async global->LDS B128 copies (ASYNCcnt path);
// V is staged transposed so the P@V B-fragment is a contiguous ds_load_b128.
// ---------------------------------------------------------------------------
#define OFF_Q   0
#define OFF_KV  66560                       // sQ: 64 x 520 bf16
#define OFF_S   (66560 + 73728)             // sK: 64x520 / sVt: 512x72 (aliased)
#define OFF_P   (66560 + 73728 + 16640)     // sS: 64 x 65 f32
#define OFF_ST  (66560 + 73728 + 16640 + 9216)  // sP: 64 x 72 bf16
#define FSH_BYTES (OFF_ST + 3 * 64 * 4)

__global__ __launch_bounds__(256) void flash64(
    const __bf16* __restrict__ Qb, const __bf16* __restrict__ Kb,
    const __bf16* __restrict__ Vb, float* __restrict__ O)
{
  extern __shared__ char smem[];
  __bf16* sQ  = (__bf16*)(smem + OFF_Q);
  __bf16* sK  = (__bf16*)(smem + OFF_KV);
  __bf16* sVt = (__bf16*)(smem + OFF_KV);   // aliased with sK
  float*  sS  = (float*)(smem + OFF_S);
  __bf16* sP  = (__bf16*)(smem + OFF_P);
  float*  sm  = (float*)(smem + OFF_ST);
  float*  sl  = sm + 64;
  float*  ssc = sl + 64;

  const int tid  = threadIdx.x;
  const int lane = tid & 31;
  const int wave = tid >> 5;                // owns O cols [wave*64, +64)
  const int q0   = blockIdx.x * 64;

  // Hardware LDS byte offset of the dynamic-LDS base for async copies.
  const unsigned int gss = __builtin_amdgcn_groupstaticsize();

  // Stage Q tile via async global->LDS (row pitch 520 bf16 = 1040 B).
  for (int c = tid; c < 4096; c += 256) {
    const int r = c >> 6, c8 = (c & 63) * 8;
    async_ld16(gss + OFF_Q + (unsigned)(r * 1040 + c8 * 2),
               Qb + (size_t)(q0 + r) * DIM + c8);
  }
  if (tid < 64) { sm[tid] = -1e30f; sl[tid] = 0.f; }

  const v8f vzero = {0.f, 0.f, 0.f, 0.f, 0.f, 0.f, 0.f, 0.f};
  v8f oacc[4][4];
#pragma unroll
  for (int i = 0; i < 4; ++i)
#pragma unroll
    for (int j = 0; j < 4; ++j) oacc[i][j] = vzero;

  for (int jb = 0; jb < L_ROWS; jb += 64) {
    __syncthreads();            // prior O-update done before K overwrites sVt
    // Stage K chunk via async global->LDS.
    for (int c = tid; c < 4096; c += 256) {
      const int r = c >> 6, c8 = (c & 63) * 8;
      async_ld16(gss + OFF_KV + (unsigned)(r * 1040 + c8 * 2),
                 Kb + (size_t)(jb + r) * DIM + c8);
    }
    wait_async0();              // Q (first iter) + K complete in this wave
    __syncthreads();

    // S = Q @ K^T : 4x4 tiles of 16x16, two tiles per wave, K-depth 512.
#pragma unroll
    for (int t = 0; t < 2; ++t) {
      const int ti = wave * 2 + t;
      const int smi = ti >> 2, sni = ti & 3;
      v8f s = vzero;
      for (int kk = 0; kk < DIM; kk += 32) {
        v16bf a = load_a_frag(&sQ[(smi * 16) * 520 + kk], 520, lane);
        v16bf b = load_b_frag(&sK[(sni * 16) * 520 + kk], 520, lane);
        s = __builtin_amdgcn_wmma_f32_16x16x32_bf16(
            false, a, false, b, (short)0, s, false, false);
      }
      const int cc = sni * 16 + (lane & 15);
      const int rb = smi * 16 + (lane >> 4) * 8;
#pragma unroll
      for (int i = 0; i < 8; ++i) sS[(rb + i) * 65 + cc] = s[i] * ATT_SCALE;
    }
    __syncthreads();

    // Online softmax: one thread per row.
    if (tid < 64) {
      const float mold = sm[tid];
      float mnew = mold;
#pragma unroll 8
      for (int c = 0; c < 64; ++c) mnew = fmaxf(mnew, sS[tid * 65 + c]);
      const float sc = __expf(mold - mnew);
      float l = sl[tid] * sc;
#pragma unroll 8
      for (int c = 0; c < 64; ++c) {
        const float p = __expf(sS[tid * 65 + c] - mnew);
        l += p;
        sP[tid * 72 + c] = (__bf16)p;
      }
      sm[tid] = mnew; sl[tid] = l; ssc[tid] = sc;
    }
    __syncthreads();

    // Rescale running O accumulators by exp(m_old - m_new).
#pragma unroll
    for (int tmi = 0; tmi < 4; ++tmi) {
#pragma unroll
      for (int i = 0; i < 8; ++i) {
        const float f = ssc[tmi * 16 + (lane >> 4) * 8 + i];
#pragma unroll
        for (int tni = 0; tni < 4; ++tni) oacc[tmi][tni][i] *= f;
      }
    }

    // Stage V chunk TRANSPOSED (d-major) into the K region: sVt[d][k].
    for (int c = tid; c < 4096; c += 256) {
      const int r = c >> 6, d8 = (c & 63) * 8;
      v8bf v = *(const v8bf*)(Vb + (size_t)(jb + r) * DIM + d8);
#pragma unroll
      for (int e = 0; e < 8; ++e) sVt[(d8 + e) * 72 + r] = v[e];
    }
    __syncthreads();

    // O += P(64x64) @ V(64 x 64-col slice); B[k,n] = sVt[col][k] contiguous.
#pragma unroll
    for (int tmi = 0; tmi < 4; ++tmi) {
#pragma unroll
      for (int tni = 0; tni < 4; ++tni) {
#pragma unroll
        for (int kk = 0; kk < 64; kk += 32) {
          v16bf a = load_a_frag(&sP[(tmi * 16) * 72 + kk], 72, lane);
          v16bf b = load_b_frag(&sVt[(wave * 64 + tni * 16) * 72 + kk], 72, lane);
          oacc[tmi][tni] = __builtin_amdgcn_wmma_f32_16x16x32_bf16(
              false, a, false, b, (short)0, oacc[tmi][tni], false, false);
        }
      }
    }
  }

  // Epilogue: divide by softmax denominator, write fp32 O.
  const int cn  = lane & 15;
  const int rlo = (lane >> 4) * 8;
#pragma unroll
  for (int tmi = 0; tmi < 4; ++tmi) {
#pragma unroll
    for (int i = 0; i < 8; ++i) {
      const int rl = tmi * 16 + rlo + i;
      const float inv = 1.0f / sl[rl];
#pragma unroll
      for (int tni = 0; tni < 4; ++tni) {
        const int col = wave * 64 + tni * 16 + cn;
        O[(size_t)(q0 + rl) * DIM + col] = oacc[tmi][tni][i] * inv;
      }
    }
  }
}

// ---------------------------------------------------------------------------
// In-place LayerNorm + ReLU over rows of X[*, 512]; one block per row.
// ---------------------------------------------------------------------------
__global__ __launch_bounds__(256) void ln_relu(
    float* __restrict__ X, const float* __restrict__ gamma,
    const float* __restrict__ beta)
{
  __shared__ float red[256];
  const int tid = threadIdx.x;
  float* x = X + (size_t)blockIdx.x * DIM;
  const float a = x[tid], b = x[tid + 256];

  red[tid] = a + b;
  __syncthreads();
  for (int s = 128; s > 0; s >>= 1) {
    if (tid < s) red[tid] += red[tid + s];
    __syncthreads();
  }
  const float mu = red[0] * (1.0f / DIM);
  __syncthreads();

  const float da = a - mu, db = b - mu;
  red[tid] = da * da + db * db;
  __syncthreads();
  for (int s = 128; s > 0; s >>= 1) {
    if (tid < s) red[tid] += red[tid + s];
    __syncthreads();
  }
  const float rs = rsqrtf(red[0] * (1.0f / DIM) + LN_EPS);

  x[tid]       = fmaxf(0.f, da * rs * gamma[tid]       + beta[tid]);
  x[tid + 256] = fmaxf(0.f, db * rs * gamma[tid + 256] + beta[tid + 256]);
}

// ---------------------------------------------------------------------------
// Workspace layout (bytes): Qb 8MB | Kb 8MB | Vb 8MB | O 16MB  (total 40MB)
// ---------------------------------------------------------------------------
extern "C" void kernel_launch(void* const* d_in, const int* in_sizes, int n_in,
                              void* d_out, int out_size, void* d_ws, size_t ws_size,
                              hipStream_t stream) {
  (void)in_sizes; (void)n_in; (void)out_size; (void)ws_size;

  const float* St = (const float*)d_in[0];
  const float* Lt = (const float*)d_in[1];
  const float* W1 = (const float*)d_in[2];
  const float* b1 = (const float*)d_in[3];
  const float* W2 = (const float*)d_in[4];
  const float* b2 = (const float*)d_in[5];
  const float* W3 = (const float*)d_in[6];
  const float* b3 = (const float*)d_in[7];
  const float* W4 = (const float*)d_in[8];
  const float* b4 = (const float*)d_in[9];
  const float* g  = (const float*)d_in[10];
  const float* be = (const float*)d_in[11];
  float* out = (float*)d_out;

  char* ws = (char*)d_ws;
  __bf16* Qb = (__bf16*)(ws);
  __bf16* Kb = (__bf16*)(ws + 8ull  * 1024 * 1024);
  __bf16* Vb = (__bf16*)(ws + 16ull * 1024 * 1024);
  float*  O  = (float*) (ws + 24ull * 1024 * 1024);

  const dim3 blk(256);
  const dim3 ggrid(DIM / 128, N_ROWS / 128);   // (4, 64)

  hipLaunchKernelGGL(gemm512, ggrid, blk, 0, stream, St, W1, b1, (const float*)nullptr, Qb, (float*)nullptr);
  hipLaunchKernelGGL(gemm512, ggrid, blk, 0, stream, Lt, W2, b2, (const float*)nullptr, Kb, (float*)nullptr);
  hipLaunchKernelGGL(gemm512, ggrid, blk, 0, stream, Lt, W3, b3, (const float*)nullptr, Vb, (float*)nullptr);
  hipLaunchKernelGGL(flash64, dim3(N_ROWS / 64), blk, FSH_BYTES, stream, Qb, Kb, Vb, O);
  hipLaunchKernelGGL(ln_relu, dim3(N_ROWS), blk, 0, stream, O, g, be);
  hipLaunchKernelGGL(gemm512, ggrid, blk, 0, stream, O, W4, b4, St, (__bf16*)nullptr, out);
}
